// GFA_22608707846585
// MI455X (gfx1250) — matrix operand order repaired and checked by reading, not compiled, not run on hardware
//
#include <hip/hip_runtime.h>
#include <hip/hip_bf16.h>

typedef __attribute__((ext_vector_type(16))) _Float16 v16h;
typedef __attribute__((ext_vector_type(8)))  float    v8f;
typedef __attribute__((ext_vector_type(4)))  unsigned int u32x4;
typedef __attribute__((ext_vector_type(8)))  unsigned int u32x8;

#define BB 16
#define NN 4096
#define CC 512
#define MM 1024   // reduced tokens (32x32)

// ---------------- WMMA fragment helpers (CDNA5 16x16x32 f16 layouts) ----------------

// A fragment 16x32 (MxK) f16 from row-major matrix, row stride lda (elements).
// lane l (l&15 = M row, l>>4 = half): halves i -> K = k0 + hi*16 + i (contiguous)
__device__ __forceinline__ v16h load_a_frag(const _Float16* A, int lda, int row0, int k0, int lane) {
    int hi = lane >> 4;
    const _Float16* p = A + (size_t)(row0 + (lane & 15)) * lda + k0 + hi * 16;
    v16h out;
#pragma unroll
    for (int i = 0; i < 8; ++i) out[i] = p[i];
#pragma unroll
    for (int i = 0; i < 8; ++i) out[8 + i] = p[8 + i];
    return out;
}

__device__ __forceinline__ v16h load_a_frag_row(const _Float16* rowp, int k0, int lane) {
    int hi = lane >> 4;
    const _Float16* p = rowp + k0 + hi * 16;
    v16h out;
#pragma unroll
    for (int i = 0; i < 8; ++i) out[i] = p[i];
#pragma unroll
    for (int i = 0; i < 8; ++i) out[8 + i] = p[8 + i];
    return out;
}

// B fragment 32x16 (KxN) sourced from W stored (N,K) row-major => computes A @ W^T.
// lane l: column N = col0 + (l&15); halves i -> K = k0 + (l>>4)*16 + i (contiguous load).
__device__ __forceinline__ v16h load_bT_frag(const _Float16* W, int ldw, int col0, int k0, int lane) {
    const _Float16* p = W + (size_t)(col0 + (lane & 15)) * ldw + k0 + (lane >> 4) * 16;
    v16h out;
#pragma unroll
    for (int i = 0; i < 16; ++i) out[i] = p[i];
    return out;
}

// B fragment 32x16 (KxN) from a row-major KxN matrix (stride ldb): strided column gather.
__device__ __forceinline__ v16h load_b_frag(const _Float16* Bm, int ldb, int k0, int col0, int lane) {
    int n  = col0 + (lane & 15);
    int kb = k0 + (lane >> 4) * 16;
    v16h out;
#pragma unroll
    for (int i = 0; i < 16; ++i) out[i] = Bm[(size_t)(kb + i) * ldb + n];
    return out;
}

__device__ __forceinline__ v8f wmma_f16(v16h a, v16h b, v8f c) {
    return __builtin_amdgcn_wmma_f32_16x16x32_f16(false, a, false, b, (short)0, c, false, false);
}

// ---------------- Tensor Data Mover: 2-D tile (rows x 32..) global -> packed LDS ----------
// D# per CDNA5 ISA ch.8: group0 = {count/lds_addr/global_addr/type}, group1 = dims/strides.
// data_size = 1 (2-byte f16). Tile rows are packed contiguously in LDS (tile_d0 elems each).
__device__ __forceinline__ void tdm_load_tile_2d(unsigned lds_byte_off,
                                                 const _Float16* gsrc,
                                                 unsigned tensor_d0, unsigned tensor_d1,
                                                 unsigned tile_d0,  unsigned tile_d1,
                                                 unsigned row_stride_elems) {
    unsigned long long ga = (unsigned long long)gsrc;
    u32x4 g0;
    g0[0] = 1u;                                        // count=1, user mode, no gather
    g0[1] = lds_byte_off;                              // lds_addr
    g0[2] = (unsigned)(ga & 0xffffffffu);              // global_addr[31:0]
    g0[3] = (unsigned)((ga >> 32) & 0x01ffffffu)       // global_addr[56:32]
            | 0x80000000u;                             // type = 2 ("image")
    u32x8 g1;
    g1[0] = 0x00010000u;                               // wg_mask=0, data_size=1 (2B)
    g1[1] = (tensor_d0 & 0xffffu) << 16;               // abar_addr=0 | tensor_dim0[15:0]
    g1[2] = (tensor_d0 >> 16) | ((tensor_d1 & 0xffffu) << 16);
    g1[3] = (tensor_d1 >> 16) | (tile_d0 << 16);       // tensor_dim1[31:16] | tile_dim0
    g1[4] = tile_d1;                                   // tile_dim1 | tile_dim2=0
    g1[5] = row_stride_elems;                          // tensor_dim0_stride[31:0]
    g1[6] = 0;                                         // stride0[47:32] | stride1[15:0]
    g1[7] = 0;                                         // tensor_dim1_stride (unused, 2-D)
    asm volatile("tensor_load_to_lds %0, %1" :: "s"(g0), "s"(g1) : "memory");
}

__device__ __forceinline__ unsigned lds_off_of(const void* p) {
    // On CDNA5 the low 32 bits of a flat shared address are the LDS byte address.
    return (unsigned)(unsigned long long)p;
}

// ---------------- elementwise / repack ----------------

__global__ void cvt_f32_f16_kernel(const float* __restrict__ in, _Float16* __restrict__ out,
                                   long n, float scale) {
    long i = (long)blockIdx.x * blockDim.x + threadIdx.x;
    long stride = (long)gridDim.x * blockDim.x;
    for (; i < n; i += stride) out[i] = (_Float16)(in[i] * scale);
}

// sr_w (O=512, I=512, R=2, S=2) f32  ->  srh[p][o][i] f16, p = r*2+s
__global__ void repack_srw_kernel(const float* __restrict__ srw, _Float16* __restrict__ srh) {
    long n = (long)CC * CC * 4;
    long i = (long)blockIdx.x * blockDim.x + threadIdx.x;
    long stride = (long)gridDim.x * blockDim.x;
    for (; i < n; i += stride) {
        int s = (int)(i & 1);
        int r = (int)((i >> 1) & 1);
        long oc = i >> 2;              // o*512 + c
        int p = r * 2 + s;
        srh[(long)p * CC * CC + oc] = (_Float16)srw[i];
    }
}

// ---------------- conv (stride-2 2x2) as 4 accumulated WMMA GEMMs ----------------
__global__ __launch_bounds__(32) void conv_kernel(const _Float16* __restrict__ xh,
                                                  const _Float16* __restrict__ srh,
                                                  const float* __restrict__ sr_b,
                                                  float* __restrict__ x2f) {
    int b    = blockIdx.z;
    int m0   = blockIdx.y * 16;
    int n0   = blockIdx.x * 16;
    int lane = threadIdx.x;

    const _Float16* X = xh + (size_t)b * NN * CC;
    int m  = m0 + (lane & 15);
    int oy = m >> 5, ox = m & 31;          // 32x32 output grid

    v8f acc = {};
#pragma unroll
    for (int p = 0; p < 4; ++p) {
        int pix = (2 * oy + (p >> 1)) * 64 + 2 * ox + (p & 1);
        const _Float16* Arow = X + (size_t)pix * CC;
        const _Float16* W    = srh + (size_t)p * CC * CC;
        for (int k0 = 0; k0 < CC; k0 += 32) {
            v16h a  = load_a_frag_row(Arow, k0, lane);
            v16h bf = load_bT_frag(W, CC, n0, k0, lane);
            acc = wmma_f16(a, bf, acc);
        }
    }
    int n = n0 + (lane & 15);
    float bias = sr_b[n];
    float* Out = x2f + (size_t)b * MM * CC;
    int rb = (lane >> 4) * 8;
#pragma unroll
    for (int r = 0; r < 8; ++r) Out[(size_t)(m0 + rb + r) * CC + n] = acc[r] + bias;
}

// ---------------- LayerNorm (one wave per 512-elem row), f32 in -> f16 out ----------------
__global__ __launch_bounds__(32) void ln_kernel(const float* __restrict__ x2f,
                                                const float* __restrict__ g,
                                                const float* __restrict__ bta,
                                                _Float16* __restrict__ x2h) {
    size_t row = blockIdx.x;
    int lane = threadIdx.x;
    const float* in = x2f + row * CC;
    float v[16];
    float s = 0.f;
#pragma unroll
    for (int i = 0; i < 16; ++i) { v[i] = in[lane * 16 + i]; s += v[i]; }
#pragma unroll
    for (int o = 16; o >= 1; o >>= 1) s += __shfl_xor(s, o, 32);
    float mu = s * (1.0f / CC);
    float vs = 0.f;
#pragma unroll
    for (int i = 0; i < 16; ++i) { float d = v[i] - mu; vs += d * d; }
#pragma unroll
    for (int o = 16; o >= 1; o >>= 1) vs += __shfl_xor(vs, o, 32);
    float inv = rsqrtf(vs * (1.0f / CC) + 1e-5f);
    _Float16* out = x2h + row * CC;
#pragma unroll
    for (int i = 0; i < 16; ++i) {
        int c = lane * 16 + i;
        out[c] = (_Float16)((v[i] - mu) * inv * g[c] + bta[c]);
    }
}

// ---------------- TDM-fed, LDS-tiled GEMM ----------------
// Out(M,N) = (A(M,K) @ W(N,K)^T + bias) * out_scale
// Block: 256 threads = 8 waves, 128x128 output tile, K stepped by 32.
// Wave 0 drives the Tensor Data Mover (double-buffered A/B tiles in LDS);
// every wave computes a 32(M) x 64(N) sub-tile = 8 WMMA accumulators.
#define KT 32
template <bool OUT_F16>
__global__ __launch_bounds__(256) void gemm_tdm_kernel(const _Float16* __restrict__ A,
                                                       const _Float16* __restrict__ W,
                                                       const float* __restrict__ bias,
                                                       void* __restrict__ Out,
                                                       int M, int N, int K, float out_scale) {
    __shared__ _Float16 sA[2][128 * KT];   // packed 128 rows x 32 K
    __shared__ _Float16 sB[2][128 * KT];   // packed 128 cols(N) x 32 K (W rows)

    int m0 = blockIdx.y * 128, n0 = blockIdx.x * 128;
    int tid  = threadIdx.x;
    int lane = tid & 31;
    int wave = tid >> 5;
    int wr   = wave >> 1;      // 0..3 -> 32-row band
    int wc   = wave & 1;       // 0..1 -> 64-col band

    const int nk = K / KT;

    // prologue: stage 0
    if (wave == 0) {
        tdm_load_tile_2d(lds_off_of(&sA[0][0]), A + (size_t)m0 * K,
                         (unsigned)K, (unsigned)M, KT, 128, (unsigned)K);
        tdm_load_tile_2d(lds_off_of(&sB[0][0]), W + (size_t)n0 * K,
                         (unsigned)K, (unsigned)N, KT, 128, (unsigned)K);
    }

    v8f acc[2][4];
#pragma unroll
    for (int rf = 0; rf < 2; ++rf)
#pragma unroll
        for (int cf = 0; cf < 4; ++cf) acc[rf][cf] = (v8f){};

    for (int ks = 0; ks < nk; ++ks) {
        int cur = ks & 1;
        if (wave == 0) {
            if (ks + 1 < nk) {
                int nxt = cur ^ 1;
                tdm_load_tile_2d(lds_off_of(&sA[nxt][0]),
                                 A + (size_t)m0 * K + (ks + 1) * KT,
                                 (unsigned)K, (unsigned)M, KT, 128, (unsigned)K);
                tdm_load_tile_2d(lds_off_of(&sB[nxt][0]),
                                 W + (size_t)n0 * K + (ks + 1) * KT,
                                 (unsigned)K, (unsigned)N, KT, 128, (unsigned)K);
                __builtin_amdgcn_s_wait_tensorcnt(2);   // stage ks complete, ks+1 in flight
            } else {
                __builtin_amdgcn_s_wait_tensorcnt(0);   // final stage complete
            }
        }
        __syncthreads();   // LDS tiles for stage ks visible to all waves

        const _Float16* tA = &sA[cur][0];
        const _Float16* tB = &sB[cur][0];
        v16h af[2], bf[4];
#pragma unroll
        for (int rf = 0; rf < 2; ++rf)
            af[rf] = load_a_frag(tA, KT, wr * 32 + rf * 16, 0, lane);
#pragma unroll
        for (int cf = 0; cf < 4; ++cf)
            bf[cf] = load_bT_frag(tB, KT, wc * 64 + cf * 16, 0, lane);
#pragma unroll
        for (int rf = 0; rf < 2; ++rf)
#pragma unroll
            for (int cf = 0; cf < 4; ++cf)
                acc[rf][cf] = wmma_f16(af[rf], bf[cf], acc[rf][cf]);

        __syncthreads();   // all waves done reading before TDM overwrites next iter
    }

    // epilogue: bias + scale + store
    int rb = (lane >> 4) * 8;
#pragma unroll
    for (int rf = 0; rf < 2; ++rf) {
#pragma unroll
        for (int cf = 0; cf < 4; ++cf) {
            int n = n0 + wc * 64 + cf * 16 + (lane & 15);
            float bv = bias[n];
            size_t rowbase = (size_t)(m0 + wr * 32 + rf * 16 + rb);
            if (OUT_F16) {
                _Float16* O = (_Float16*)Out;
#pragma unroll
                for (int r = 0; r < 8; ++r)
                    O[(rowbase + r) * N + n] = (_Float16)((acc[rf][cf][r] + bv) * out_scale);
            } else {
                float* O = (float*)Out;
#pragma unroll
                for (int r = 0; r < 8; ++r)
                    O[(rowbase + r) * N + n] = (acc[rf][cf][r] + bv) * out_scale;
            }
        }
    }
}

// ---------------- attention: one 16-query tile per workgroup (4 waves) ----------------
__global__ __launch_bounds__(128) void attention_kernel(const _Float16* __restrict__ qh,
                                                        const _Float16* __restrict__ kvh,
                                                        _Float16* __restrict__ outh) {
    __shared__ _Float16 sP[16 * 1024];   // scores then probs, f16
    __shared__ float red[16][8];
    __shared__ float rowmax[16];
    __shared__ float rowinv[16];

    int b    = blockIdx.y;
    int qt   = blockIdx.x;               // 0..255
    int tid  = threadIdx.x;
    int lane = tid & 31;
    int wave = tid >> 5;

    const _Float16* Q = qh  + (size_t)b * NN * CC + (size_t)qt * 16 * CC;   // 16 x 512
    const _Float16* K = kvh + (size_t)b * MM * (2 * CC);                    // keys, stride 1024
    const _Float16* V = K + CC;                                             // values, stride 1024

    for (int kt = wave; kt < 64; kt += 4) {
        v8f acc = {};
        for (int k0 = 0; k0 < CC; k0 += 32) {
            v16h a  = load_a_frag(Q, CC, 0, k0, lane);
            v16h bf = load_bT_frag(K, 2 * CC, kt * 16, k0, lane);
            acc = wmma_f16(a, bf, acc);
        }
        int n  = kt * 16 + (lane & 15);
        int rb = (lane >> 4) * 8;
#pragma unroll
        for (int r = 0; r < 8; ++r) sP[(rb + r) * 1024 + n] = (_Float16)acc[r];
    }
    __syncthreads();

    {
        int row = tid >> 3, seg = tid & 7;
        float m = -1e30f;
        for (int j = seg * 128; j < seg * 128 + 128; ++j)
            m = fmaxf(m, (float)sP[row * 1024 + j]);
        red[row][seg] = m;
    }
    __syncthreads();
    if (tid < 16) {
        float m = red[tid][0];
#pragma unroll
        for (int s = 1; s < 8; ++s) m = fmaxf(m, red[tid][s]);
        rowmax[tid] = m;
    }
    __syncthreads();
    {
        int row = tid >> 3, seg = tid & 7;
        float m = rowmax[row], s = 0.f;
        for (int j = seg * 128; j < seg * 128 + 128; ++j)
            s += __expf((float)sP[row * 1024 + j] - m);
        red[row][seg] = s;
    }
    __syncthreads();
    if (tid < 16) {
        float s = 0.f;
#pragma unroll
        for (int e = 0; e < 8; ++e) s += red[tid][e];
        rowinv[tid] = 1.0f / s;
    }
    __syncthreads();
    {
        int row = tid >> 3, seg = tid & 7;
        float m = rowmax[row], inv = rowinv[row];
        for (int j = seg * 128; j < seg * 128 + 128; ++j) {
            float p = __expf((float)sP[row * 1024 + j] - m) * inv;
            sP[row * 1024 + j] = (_Float16)p;
        }
    }
    __syncthreads();

    _Float16* O = outh + (size_t)b * NN * CC + (size_t)qt * 16 * CC;
    for (int ct = wave; ct < 32; ct += 4) {
        v8f acc = {};
        for (int k0 = 0; k0 < 1024; k0 += 32) {
            v16h a  = load_a_frag(sP, 1024, 0, k0, lane);
            v16h bf = load_b_frag(V, 2 * CC, k0, ct * 16, lane);
            acc = wmma_f16(a, bf, acc);
        }
        int n  = ct * 16 + (lane & 15);
        int rb = (lane >> 4) * 8;
#pragma unroll
        for (int r = 0; r < 8; ++r) O[(size_t)(rb + r) * CC + n] = (_Float16)acc[r];
    }
}

// ---------------- host orchestration ----------------

extern "C" void kernel_launch(void* const* d_in, const int* in_sizes, int n_in,
                              void* d_out, int out_size, void* d_ws, size_t ws_size,
                              hipStream_t stream) {
    (void)in_sizes; (void)n_in; (void)out_size; (void)ws_size;

    const float* x      = (const float*)d_in[0];
    const float* sr_w   = (const float*)d_in[1];
    const float* sr_b   = (const float*)d_in[2];
    const float* ln_g   = (const float*)d_in[3];
    const float* ln_b   = (const float*)d_in[4];
    const float* q_w    = (const float*)d_in[5];
    const float* q_b    = (const float*)d_in[6];
    const float* kv_w   = (const float*)d_in[7];
    const float* kv_b   = (const float*)d_in[8];
    const float* proj_w = (const float*)d_in[9];
    const float* proj_b = (const float*)d_in[10];

    const long nx   = (long)BB * NN * CC;       // 33.5M
    const long nx2  = (long)BB * MM * CC;       // 8.4M

    char* ws = (char*)d_ws;
    size_t off = 0;
    auto take = [&](size_t bytes) -> char* {
        char* p = ws + off;
        off += (bytes + 255) & ~(size_t)255;
        return p;
    };
    _Float16* xh      = (_Float16*)take(nx * 2);                 // x f16; reused as outh later
    _Float16* qh      = (_Float16*)take(nx * 2);                 // q f16 (pre-scaled)
    _Float16* kvh     = (_Float16*)take((size_t)BB * MM * 2 * CC * 2);
    float*    x2f     = (float*)   take(nx2 * 4);
    _Float16* x2h     = (_Float16*)take(nx2 * 2);
    _Float16* q_wh    = (_Float16*)take((size_t)CC * CC * 2);
    _Float16* kv_wh   = (_Float16*)take((size_t)2 * CC * CC * 2);
    _Float16* proj_wh = (_Float16*)take((size_t)CC * CC * 2);
    _Float16* srh     = (_Float16*)take((size_t)4 * CC * CC * 2);

    const float scale = 0.044194173824159216f;   // 512^-0.5

    // 1) conversions + repack
    cvt_f32_f16_kernel<<<4096, 256, 0, stream>>>(x, xh, nx, 1.0f);
    cvt_f32_f16_kernel<<<512, 256, 0, stream>>>(q_w, q_wh, (long)CC * CC, 1.0f);
    cvt_f32_f16_kernel<<<1024, 256, 0, stream>>>(kv_w, kv_wh, (long)2 * CC * CC, 1.0f);
    cvt_f32_f16_kernel<<<512, 256, 0, stream>>>(proj_w, proj_wh, (long)CC * CC, 1.0f);
    repack_srw_kernel<<<2048, 256, 0, stream>>>(sr_w, srh);

    // 2) spatial-reduction conv -> x2f (f32)
    {
        dim3 g(CC / 16, MM / 16, BB);
        conv_kernel<<<g, 32, 0, stream>>>(xh, srh, sr_b, x2f);
    }

    // 3) layernorm -> x2h (f16)
    ln_kernel<<<BB * MM, 32, 0, stream>>>(x2f, ln_g, ln_b, x2h);

    // 4) q = (x @ q_w^T + q_b) * scale  -> qh (f16)   [TDM-tiled]
    {
        dim3 g(CC / 128, (BB * NN) / 128);
        gemm_tdm_kernel<true><<<g, 256, 0, stream>>>(xh, q_wh, q_b, qh,
                                                     BB * NN, CC, CC, scale);
    }

    // 5) kv = x2 @ kv_w^T + kv_b -> kvh (f16), k = cols[0:512), v = cols[512:1024)
    {
        dim3 g((2 * CC) / 128, (BB * MM) / 128);
        gemm_tdm_kernel<true><<<g, 256, 0, stream>>>(x2h, kv_wh, kv_b, kvh,
                                                     BB * MM, 2 * CC, CC, 1.0f);
    }

    // 6) attention -> outh (reuse xh buffer; x no longer needed)
    _Float16* outh = xh;
    {
        dim3 g(NN / 16, BB);
        attention_kernel<<<g, 128, 0, stream>>>(qh, kvh, outh);
    }

    // 7) final projection -> d_out (f32)   [TDM-tiled]
    {
        dim3 g(CC / 128, (BB * NN) / 128);
        gemm_tdm_kernel<false><<<g, 256, 0, stream>>>(outh, proj_wh, proj_b, d_out,
                                                      BB * NN, CC, CC, 1.0f);
    }
}
